// SelfAttentionRoPE_86122684220042
// MI455X (gfx1250) — matrix operand order repaired and checked by reading, not compiled, not run on hardware
//
#include <hip/hip_runtime.h>
#include <hip/hip_bf16.h>
#include <math.h>

// ---------------------------------------------------------------------------
// Problem dims (fixed by the harness reference)
// ---------------------------------------------------------------------------
constexpr int Bc = 2;
constexpr int Lc = 2048;
constexpr int Dc = 1024;
constexpr int Hc = 16;
constexpr int Mrows = Bc * Lc;  // 4096

// ---------------------------------------------------------------------------
// CDNA5 (gfx1250) wave32 WMMA types / helpers
// ---------------------------------------------------------------------------
typedef __attribute__((ext_vector_type(16))) __bf16 v16bf;
typedef __attribute__((ext_vector_type(8)))  float  v8f;

// D(16x16 f32) = A(16x32 bf16) x B(32x16 bf16) + C
__device__ __forceinline__ v8f wmma_bf16(v16bf a, v16bf b, v8f c) {
  return __builtin_amdgcn_wmma_f32_16x16x32_bf16(
      /*neg_a=*/false, a, /*neg_b=*/false, b,
      /*c_mod=*/(short)0, c, /*reuse_a=*/false, /*reuse_b=*/false);
}

// Load a 16-element bf16 fragment from two 16-byte contiguous chunks.
// A-fragment (16x32, per-lane row M=lane&15): chunks at K-offsets (lh*16, lh*16+8)
// B-fragment (32x16, per-lane col N=lane&15): one 32B chunk at K-offset lh*16 -> (p, p+8)
__device__ __forceinline__ v16bf ld_frag(const __bf16* p0, const __bf16* p1) {
  union { uint4 u[2]; v16bf v; } f;
  f.u[0] = *reinterpret_cast<const uint4*>(p0);
  f.u[1] = *reinterpret_cast<const uint4*>(p1);
  return f.v;
}

// ---------------------------------------------------------------------------
// fp32 -> bf16 convert
// ---------------------------------------------------------------------------
__global__ void cvt_f32_bf16(const float* __restrict__ in, __bf16* __restrict__ out, int n) {
  int i = blockIdx.x * blockDim.x + threadIdx.x;
  if (i < n) out[i] = (__bf16)in[i];
}

// ---------------------------------------------------------------------------
// GEMM: C[M,N] (f32) = A[M,K] (bf16, row-major) x W[N,K]^T (bf16, row-major)
// Compile-time N,K so all strides fold into instruction immediates.
// Block = 128 threads = 4 waves; wave tile 32x64 (2x4 WMMA), block tile 64x128.
// ---------------------------------------------------------------------------
template <int N, int K>
__global__ __launch_bounds__(128) void gemm_bf16_nt(
    const __bf16* __restrict__ A, const __bf16* __restrict__ W,
    float* __restrict__ C) {
  const int lane = threadIdx.x & 31;
  const int wid  = threadIdx.x >> 5;
  const int lr = lane & 15;        // row/col within 16x16 tile
  const int lh = lane >> 4;        // half-wave select
  const int m0 = blockIdx.x * 64 + (wid & 1) * 32;
  const int n0 = blockIdx.y * 128 + (wid >> 1) * 64;

  v8f acc[2][4];
#pragma unroll
  for (int i = 0; i < 2; ++i)
#pragma unroll
    for (int j = 0; j < 4; ++j) acc[i][j] = {};

  const __bf16* ap = A + (size_t)(m0 + lr) * K + lh * 16;       // +16*i*K per i
  const __bf16* wp = W + (size_t)(n0 + lr) * K + lh * 16;       // +16*j*K per j

  for (int k = 0; k < K; k += 32) {
    v16bf af[2], bf[4];
#pragma unroll
    for (int i = 0; i < 2; ++i) {
      const __bf16* p = ap + (size_t)(16 * i) * K + k;
      __builtin_prefetch(p + 64, 0, 1);          // -> global_prefetch_b8
      af[i] = ld_frag(p, p + 8);
    }
#pragma unroll
    for (int j = 0; j < 4; ++j) {
      const __bf16* p = wp + (size_t)(16 * j) * K + k;
      __builtin_prefetch(p + 64, 0, 1);
      bf[j] = ld_frag(p, p + 8);
    }
#pragma unroll
    for (int i = 0; i < 2; ++i)
#pragma unroll
      for (int j = 0; j < 4; ++j)
        acc[i][j] = wmma_bf16(af[i], bf[j], acc[i][j]);
  }

  // C/D layout: slot r, lane -> row = r + 8*lh, col = lr.
  // Constexpr N => per-tile base + immediate row offsets.
#pragma unroll
  for (int i = 0; i < 2; ++i)
#pragma unroll
    for (int j = 0; j < 4; ++j) {
      float* cp = C + (size_t)(m0 + 16 * i + 8 * lh) * N + n0 + 16 * j + lr;
#pragma unroll
      for (int r = 0; r < 8; ++r) cp[(size_t)r * N] = acc[i][j][r];
    }
}

// ---------------------------------------------------------------------------
// Fused RMSNorm + RoPE + layout transform.
// qkv: [B*L, 3*H*64] fp32 (e = s*1024 + h*64 + d).
// One wave per (b,l,h); lane handles rope pair (d0=2*lane, d1=2*lane+1).
// Outputs: qh,kh [B,H,L,64] bf16 ; vT [B,H,64,L] bf16 (V transposed).
// ---------------------------------------------------------------------------
__global__ void qkv_post(const float* __restrict__ qkv, const float* __restrict__ rope,
                         const float* __restrict__ qw, const float* __restrict__ kw,
                         __bf16* __restrict__ qh, __bf16* __restrict__ kh,
                         __bf16* __restrict__ vT) {
  const int lane = threadIdx.x & 31;
  const int w = (blockIdx.x * blockDim.x + threadIdx.x) >> 5;
  const int h = w % Hc;
  const int l = (w / Hc) % Lc;
  const int b = w / (Hc * Lc);
  const int d0 = lane * 2, d1 = d0 + 1;

  const float* row = qkv + (size_t)(b * Lc + l) * (3 * Dc);
  float q0 = row[h * 64 + d0],              q1 = row[h * 64 + d1];
  float k0 = row[Dc + h * 64 + d0],         k1 = row[Dc + h * 64 + d1];
  float v0 = row[2 * Dc + h * 64 + d0],     v1 = row[2 * Dc + h * 64 + d1];

  float sq = q0 * q0 + q1 * q1;
  float sk = k0 * k0 + k1 * k1;
#pragma unroll
  for (int m = 16; m >= 1; m >>= 1) {
    sq += __shfl_xor(sq, m, 32);
    sk += __shfl_xor(sk, m, 32);
  }
  float qi = rsqrtf(sq * (1.0f / 64.0f) + 1e-6f);
  float ki = rsqrtf(sk * (1.0f / 64.0f) + 1e-6f);
  q0 *= qi * qw[d0]; q1 *= qi * qw[d1];
  k0 *= ki * kw[d0]; k1 *= ki * kw[d1];

  float c = rope[(size_t)(l * 32 + lane) * 2 + 0];
  float s = rope[(size_t)(l * 32 + lane) * 2 + 1];
  float qr0 = q0 * c - q1 * s, qr1 = q0 * s + q1 * c;
  float kr0 = k0 * c - k1 * s, kr1 = k0 * s + k1 * c;

  const size_t bh = (size_t)(b * Hc + h);
  size_t qkb = (bh * Lc + l) * 64;
  qh[qkb + d0] = (__bf16)qr0; qh[qkb + d1] = (__bf16)qr1;
  kh[qkb + d0] = (__bf16)kr0; kh[qkb + d1] = (__bf16)kr1;
  vT[(bh * 64 + d0) * Lc + l] = (__bf16)v0;
  vT[(bh * 64 + d1) * Lc + l] = (__bf16)v1;
}

// ---------------------------------------------------------------------------
// Flash attention (no mask), one wave per 16-query tile.
// Computes S^T = K·Q^T so softmax stats live per-lane (query = lane&15),
// then O^T += V^T · P^T with P^T moved C-layout -> B-fragment via bpermute.
// Output written bf16 as [B*L, H*64] for the projection GEMM.
// ---------------------------------------------------------------------------
__global__ __launch_bounds__(128) void attn_fwd(
    const __bf16* __restrict__ qh, const __bf16* __restrict__ kh,
    const __bf16* __restrict__ vT, __bf16* __restrict__ out) {
  const int lane  = threadIdx.x & 31;
  const int wtile = blockIdx.x * 4 + (threadIdx.x >> 5);
  constexpr int nqt = Lc / 16;
  const int qt = wtile % nqt;
  const int h  = (wtile / nqt) % Hc;
  const int b  = wtile / (nqt * Hc);
  const int lr = lane & 15;
  const int lh = lane >> 4;

  const size_t bh = (size_t)(b * Hc + h);
  const __bf16* Q  = qh + (bh * Lc + (size_t)qt * 16) * 64;
  const __bf16* Kp = kh + bh * Lc * 64;
  const __bf16* Vt = vT + bh * 64 * Lc;

  // Q^T B-fragments (N = query = lr), 2 frags cover Dh=64
  v16bf qf[2];
#pragma unroll
  for (int f = 0; f < 2; ++f) {
    const __bf16* p = Q + (size_t)lr * 64 + f * 32 + lh * 16;
    qf[f] = ld_frag(p, p + 8);
  }

  v8f o[4];
  o[0] = {}; o[1] = {}; o[2] = {}; o[3] = {};
  float mrun = -__builtin_inff();
  float lrun = 0.0f;

  for (int kc = 0; kc < Lc; kc += 32) {
    // S^T tiles: s0 = keys kc..kc+15, s1 = keys kc+16..kc+31 (rows); queries (cols)
    v8f s0 = {}, s1 = {};
#pragma unroll
    for (int f = 0; f < 2; ++f) {
      const __bf16* kp0 = Kp + (size_t)(kc + lr) * 64 + f * 32 + lh * 16;
      s0 = wmma_bf16(ld_frag(kp0, kp0 + 8), qf[f], s0);
      const __bf16* kp1 = Kp + (size_t)(kc + 16 + lr) * 64 + f * 32 + lh * 16;
      s1 = wmma_bf16(ld_frag(kp1, kp1 + 8), qf[f], s1);
    }

    // scale + online softmax; lane owns query lr, keys {8*lh..} U {16+8*lh..}
    float mloc = -__builtin_inff();
#pragma unroll
    for (int r = 0; r < 8; ++r) {
      s0[r] *= 0.125f;  // 1/sqrt(64)
      s1[r] *= 0.125f;
      mloc = fmaxf(mloc, fmaxf(s0[r], s1[r]));
    }
    mloc = fmaxf(mloc, __shfl_xor(mloc, 16, 32));  // combine key-halves
    float mnew  = fmaxf(mrun, mloc);
    float alpha = __expf(mrun - mnew);

    v8f p0, p1;
    float psum = 0.0f;
#pragma unroll
    for (int r = 0; r < 8; ++r) {
      p0[r] = __expf(s0[r] - mnew);
      p1[r] = __expf(s1[r] - mnew);
      psum += p0[r] + p1[r];
    }
    psum += __shfl_xor(psum, 16, 32);
    lrun = lrun * alpha + psum;
    mrun = mnew;

#pragma unroll
    for (int m = 0; m < 4; ++m)
#pragma unroll
      for (int r = 0; r < 8; ++r)
        o[m][r] *= alpha;

    // Build P^T B-fragment: elem j @ lane = P^T[key=16*lh+j, query=lr]
    // source: tile (lh), slot (j&7), lane 16*(j>>3)+lr
    v16bf pf;
#pragma unroll
    for (int j = 0; j < 16; ++j) {
      int sl = ((j >> 3) << 4) + lr;
      float a0 = __shfl(p0[j & 7], sl, 32);
      float a1 = __shfl(p1[j & 7], sl, 32);
      pf[j] = (__bf16)((lane < 16) ? a0 : a1);
    }

    // O^T(16d x 16q) += V^T(16d x 32k) · P^T(32k x 16q), 4 d-tiles
#pragma unroll
    for (int m = 0; m < 4; ++m) {
      const __bf16* vp = Vt + (size_t)(m * 16 + lr) * Lc + kc + lh * 16;
      o[m] = wmma_bf16(ld_frag(vp, vp + 8), pf, o[m]);
    }
  }

  // normalize + store: lane holds (d = 16m + r + 8*lh, query = lr)
  float inv = 1.0f / lrun;
  __bf16* orow = out + ((size_t)b * Lc + (size_t)qt * 16 + lr) * Dc + h * 64;
#pragma unroll
  for (int m = 0; m < 4; ++m)
#pragma unroll
    for (int r = 0; r < 8; ++r)
      orow[m * 16 + r + 8 * lh] = (__bf16)(o[m][r] * inv);
}

// ---------------------------------------------------------------------------
// Launch
// ---------------------------------------------------------------------------
extern "C" void kernel_launch(void* const* d_in, const int* in_sizes, int n_in,
                              void* d_out, int out_size, void* d_ws, size_t ws_size,
                              hipStream_t stream) {
  (void)in_sizes; (void)n_in; (void)out_size; (void)ws_size;

  const float* x      = (const float*)d_in[0];
  const float* rope   = (const float*)d_in[1];
  const float* w_qkv  = (const float*)d_in[2];
  const float* w_proj = (const float*)d_in[3];
  const float* q_w    = (const float*)d_in[4];
  const float* k_w    = (const float*)d_in[5];

  char* ws = (char*)d_ws;
  size_t off = 0;
  auto alloc = [&](size_t bytes) -> void* {
    void* p = ws + off;
    off += (bytes + 255) & ~(size_t)255;
    return p;
  };
  __bf16* x_bf     = (__bf16*)alloc((size_t)Mrows * Dc * 2);        //  8.4 MB
  __bf16* wqkv_bf  = (__bf16*)alloc((size_t)3 * Dc * Dc * 2);       //  6.3 MB
  __bf16* wproj_bf = (__bf16*)alloc((size_t)Dc * Dc * 2);           //  2.1 MB
  float*  qkv      = (float*) alloc((size_t)Mrows * 3 * Dc * 4);    // 50.3 MB
  __bf16* qh       = (__bf16*)alloc((size_t)Mrows * Dc * 2);
  __bf16* kh       = (__bf16*)alloc((size_t)Mrows * Dc * 2);
  __bf16* vT       = (__bf16*)alloc((size_t)Mrows * Dc * 2);
  __bf16* abf      = (__bf16*)alloc((size_t)Mrows * Dc * 2);

  cvt_f32_bf16<<<(Mrows * Dc) / 256, 256, 0, stream>>>(x, x_bf, Mrows * Dc);
  cvt_f32_bf16<<<(3 * Dc * Dc) / 256, 256, 0, stream>>>(w_qkv, wqkv_bf, 3 * Dc * Dc);
  cvt_f32_bf16<<<(Dc * Dc) / 256, 256, 0, stream>>>(w_proj, wproj_bf, Dc * Dc);

  // qkv[M, 3D] = x[M, D] · w_qkv[3D, D]^T
  gemm_bf16_nt<3 * Dc, Dc><<<dim3(Mrows / 64, (3 * Dc) / 128), 128, 0, stream>>>(
      x_bf, wqkv_bf, qkv);

  // rmsnorm + rope + transpose V
  qkv_post<<<(Bc * Lc * Hc * 32) / 128, 128, 0, stream>>>(qkv, rope, q_w, k_w, qh, kh, vT);

  // flash attention: one wave per 16 queries
  attn_fwd<<<(Bc * Hc * (Lc / 16)) / 4, 128, 0, stream>>>(qh, kh, vT, abf);

  // out[M, D] = attn[M, D] · w_proj[D, D]^T  (fp32 directly into d_out)
  gemm_bf16_nt<Dc, Dc><<<dim3(Mrows / 64, Dc / 128), 128, 0, stream>>>(
      abf, wproj_bf, (float*)d_out);
}
